// ContextEncoder_5411658793152
// MI455X (gfx1250) — matrix operand order repaired and checked by reading, not compiled, not run on hardware
//
#include <hip/hip_runtime.h>
#include <math.h>

// ---------------------------------------------------------------------------
// Types for CDNA5 WMMA (wave32, 16x16x32 bf16 -> f32)
// ---------------------------------------------------------------------------
typedef __bf16 bf16;
typedef __attribute__((ext_vector_type(8)))  __bf16 v8bf;
typedef __attribute__((ext_vector_type(16))) __bf16 v16bf;
typedef __attribute__((ext_vector_type(8)))  float  v8f;
typedef __attribute__((address_space(3))) bf16 lds_bf16;

#define B_    2
#define T_    2048
#define D_    512
#define H_    8
#define DH_   64
#define WCTX_ 32
#define WIN_  65
#define MTOT_ (B_ * T_)      // 4096 token rows

__device__ __forceinline__ bf16 f2b(float f) { return (bf16)f; }

// ---------------------------------------------------------------------------
// Async DMA: global (16B) -> LDS, tracked by ASYNCcnt (CDNA5 ISA).
// GV addressing mode: 64-bit VGPR global address, LDS byte offset in VGPR.
// ---------------------------------------------------------------------------
__device__ __forceinline__ void async_copy_b128(const bf16* gsrc, bf16* ldst) {
  uint64_t ga = (uint64_t)(uintptr_t)gsrc;
  uint32_t lo = (uint32_t)(uintptr_t)(lds_bf16*)ldst;
  asm volatile("global_load_async_to_lds_b128 %0, %1, off"
               :: "v"(lo), "v"(ga) : "memory");
}
__device__ __forceinline__ void wait_async_le1() {
  asm volatile("s_wait_asynccnt 0x1" ::: "memory");
}
__device__ __forceinline__ void wait_async_le0() {
  asm volatile("s_wait_asynccnt 0x0" ::: "memory");
}

// ---------------------------------------------------------------------------
// Fragment helpers (per ISA 7.12.2 VGPR layouts, wave32)
// A 16x32 bf16: lanes 0-15 -> M=lane, K = {k0..k0+7, k0+16..k0+23}
//               lanes16-31 -> M=lane-16, K = {k0+8..k0+15, k0+24..k0+31}
// B 32x16 bf16: lanes 0-15 -> N=lane, K = {k0..k0+7, k0+16..k0+23}
//               lanes16-31 -> N=lane-16, K = {k0+8..k0+15, k0+24..k0+31}
// ---------------------------------------------------------------------------
// B fragment from a precomputed per-lane row pointer (wlane already includes
// the n-row and the lane-half k-offset); k0 advances by 32 per panel.
__device__ __forceinline__ v16bf load_b_at(const bf16* wlane, int k0) {
  const v8bf* p = (const v8bf*)(wlane + k0);
  v8bf lo = p[0];
  v8bf hi = p[2];  // +16 k-elements
  return __builtin_shufflevector(lo, hi, 0, 1, 2, 3, 4, 5, 6, 7,
                                 8, 9, 10, 11, 12, 13, 14, 15);
}

// A fragment out of an LDS panel (64 rows x 32 k, row stride 32 elements)
__device__ __forceinline__ v16bf load_a_frag_lds(const bf16* sbuf, int row0,
                                                 int lane) {
  int r    = row0 + (lane & 15);
  int koff = (lane >> 4) << 3;
  const v8bf* p = (const v8bf*)(sbuf + r * 32 + koff);
  v8bf lo = p[0];
  v8bf hi = p[2];  // +16 k-elements
  return __builtin_shufflevector(lo, hi, 0, 1, 2, 3, 4, 5, 6, 7,
                                 8, 9, 10, 11, 12, 13, 14, 15);
}

// ---------------------------------------------------------------------------
// fp32 -> bf16 conversion (weights)
// ---------------------------------------------------------------------------
__global__ void cvt_kernel(const float* __restrict__ in, bf16* __restrict__ out,
                           int n) {
  int i = blockIdx.x * blockDim.x + threadIdx.x;
  if (i < n) out[i] = f2b(in[i]);
}

// ---------------------------------------------------------------------------
// LayerNorm: one wave per 512-wide row. Emits xn (bf16) and x (bf16 copy).
// ---------------------------------------------------------------------------
__global__ __launch_bounds__(256) void ln_kernel(
    const float* __restrict__ x, const float* __restrict__ g,
    const float* __restrict__ b, bf16* __restrict__ xnb,
    bf16* __restrict__ xb) {
  int lane = threadIdx.x & 31;
  int row  = blockIdx.x * 8 + (threadIdx.x >> 5);
  const float* xr = x + (size_t)row * D_;
  float vals[16];
  float s = 0.f, s2 = 0.f;
#pragma unroll
  for (int j = 0; j < 16; ++j) {
    float t = xr[lane + 32 * j];
    vals[j] = t;
    s += t;
    s2 += t * t;
  }
#pragma unroll
  for (int o = 16; o > 0; o >>= 1) {
    s  += __shfl_xor(s, o, 32);
    s2 += __shfl_xor(s2, o, 32);
  }
  float mu   = s * (1.f / (float)D_);
  float var  = s2 * (1.f / (float)D_) - mu * mu;
  float rstd = rsqrtf(var + 1e-5f);
#pragma unroll
  for (int j = 0; j < 16; ++j) {
    int c    = lane + 32 * j;
    float xn = (vals[j] - mu) * rstd * g[c] + b[c];
    xnb[(size_t)row * D_ + c] = f2b(xn);
    xb [(size_t)row * D_ + c] = f2b(vals[j]);
  }
}

// ---------------------------------------------------------------------------
// WMMA GEMM with async-LDS staged, double-buffered A panels and
// register-pipelined B fragments.
// C[M,N] = A[M,K] @ W[N,K]^T + bias (epilogues below).
// Block (8 waves) -> 64(M) x 128(N) tile; wave -> 64 x 16 strip.
// k-panel: 64 rows x 32 k bf16 = 4KB, DMA'd by all 256 threads (16B each)
// via global_load_async_to_lds_b128; panel i+1 DMA + B-frag load for panel
// i+1 overlap the WMMA burst on panel i. Last iteration is peeled so the
// steady-state loop is branch-free.
// Split-A supported for the gate concat (A0 for k<ksplit, A1 after).
// ---------------------------------------------------------------------------
enum { EPI_BF16 = 0, EPI_SCALE_F32 = 1, EPI_F32 = 2, EPI_BOTH = 3, EPI_GATE = 4 };

template <int EPI>
__global__ __launch_bounds__(256) void gemm_bf16_kernel(
    const bf16* __restrict__ A0, const bf16* __restrict__ A1, int ksplit,
    int lda, const bf16* __restrict__ W, const float* __restrict__ bias,
    float scale, int M, int N, int K, float* __restrict__ outF,
    bf16* __restrict__ outB, const float* __restrict__ xres,
    const float* __restrict__ cres) {
  __shared__ bf16 sA[2][64 * 32];

  int tid  = threadIdx.x;
  int lane = tid & 31;
  int wid  = tid >> 5;
  int nblk = N >> 7;                 // N / 128
  int m0   = (blockIdx.x / nblk) * 64;
  int n0   = (blockIdx.x % nblk) * 128 + wid * 16;

  // per-thread DMA slot: 16 bytes of a 64x32 panel
  int arow = tid >> 2;               // 0..63
  int aseg = (tid & 3) << 3;         // element offset 0,8,16,24

  // hoisted per-thread/per-lane base pointers
  const bf16* ga0   = A0 + (size_t)(m0 + arow) * lda + aseg;
  const bf16* ga1   = A1 + (size_t)(m0 + arow) * lda + aseg;
  bf16*       lds0  = &sA[0][arow * 32 + aseg];
  bf16*       lds1  = &sA[1][arow * 32 + aseg];
  const bf16* wlane = W + (size_t)(n0 + (lane & 15)) * K + ((lane >> 4) << 3);

  const int NK = K >> 5;
  v8f acc[4];
#pragma unroll
  for (int i = 0; i < 4; ++i)
    acc[i] = (v8f){0.f, 0.f, 0.f, 0.f, 0.f, 0.f, 0.f, 0.f};

  // issue async DMA for panel p into buffer p&1
  auto issue = [&](int p) {
    int kbase = p << 5;
    const bf16* g =
        (kbase < ksplit) ? (ga0 + kbase) : (ga1 + (kbase - ksplit));
    async_copy_b128(g, (p & 1) ? lds1 : lds0);
  };

  // 4 LDS A-fragments + 4 WMMAs on one resident panel
  auto compute = [&](int it, v16bf bcur) {
    const bf16* sbuf = sA[it & 1];
    v16bf af[4];
#pragma unroll
    for (int i = 0; i < 4; ++i) af[i] = load_a_frag_lds(sbuf, 16 * i, lane);
#pragma unroll
    for (int i = 0; i < 4; ++i)
      acc[i] = __builtin_amdgcn_wmma_f32_16x16x32_bf16(
          false, af[i], false, bcur, (short)0, acc[i], false, false);
  };

  issue(0);
  v16bf bcur = load_b_at(wlane, 0);
  for (int it = 0; it < NK - 1; ++it) {
    issue(it + 1);                          // DMA next A panel
    v16bf bnext = load_b_at(wlane, (it + 1) << 5);  // prefetch next B frag
    wait_async_le1();                       // panel `it` complete (in-order)
    __syncthreads();                        // panel visible to all waves
    compute(it, bcur);
    __syncthreads();                        // reads done before buffer reuse
    bcur = bnext;
  }
  wait_async_le0();
  __syncthreads();
  compute(NK - 1, bcur);

  int n     = n0 + (lane & 15);
  float bs  = bias[n];
  int mbase = m0 + ((lane >> 4) << 3);
#pragma unroll
  for (int i = 0; i < 4; ++i) {
#pragma unroll
    for (int j = 0; j < 8; ++j) {
      int m      = mbase + 16 * i + j;
      float val  = acc[i][j] + bs;
      size_t idx = (size_t)m * N + n;
      if (EPI == EPI_BF16) {
        outB[idx] = f2b(val);
      } else if (EPI == EPI_SCALE_F32) {
        outF[idx] = val * scale;
      } else if (EPI == EPI_F32) {
        outF[idx] = val;
      } else if (EPI == EPI_BOTH) {
        outF[idx] = val;
        outB[idx] = f2b(val);
      } else {  // EPI_GATE: out = x*(1-g) + ctx*g, g = sigmoid(val)
        float gte = 1.f / (1.f + __expf(-val));
        outF[idx] = xres[idx] * (1.f - gte) + cres[idx] * gte;
      }
    }
  }
}

// ---------------------------------------------------------------------------
// Banded attention. Block = 128 threads, one (b, h, 128-query strip).
// k/v window rows staged to LDS (bf16). One thread per query; q + output
// accumulator live in registers; online softmax over the 65-wide band.
// ---------------------------------------------------------------------------
#define ATT_TB   128
#define ATT_ROWS (ATT_TB + 2 * WCTX_)  // 192

__global__ __launch_bounds__(ATT_TB) void attn_kernel(
    const float* __restrict__ q, const float* __restrict__ k,
    const float* __restrict__ v, bf16* __restrict__ out) {
  __shared__ bf16 sk[ATT_ROWS * DH_];
  __shared__ bf16 sv[ATT_ROWS * DH_];

  int tid = threadIdx.x;
  int t0  = blockIdx.x * ATT_TB;
  int h   = blockIdx.y;
  int b   = blockIdx.z;

  for (int i = tid; i < ATT_ROWS * DH_; i += ATT_TB) {
    int r  = i >> 6;
    int c  = i & 63;
    int tt = t0 - WCTX_ + r;
    float kv = 0.f, vv = 0.f;
    if (tt >= 0 && tt < T_) {
      size_t idx = ((size_t)(b * T_ + tt)) * D_ + h * DH_ + c;
      kv = k[idx];
      vv = v[idx];
    }
    sk[i] = f2b(kv);
    sv[i] = f2b(vv);
  }
  __syncthreads();

  int t = t0 + tid;
  float qr[DH_];
  {
    const float* qp = q + ((size_t)(b * T_ + t)) * D_ + h * DH_;
#pragma unroll
    for (int d = 0; d < DH_; ++d) qr[d] = qp[d];
  }

  float mx = -1e30f, lsum = 0.f;
  float o[DH_];
#pragma unroll
  for (int d = 0; d < DH_; ++d) o[d] = 0.f;

#pragma unroll 1
  for (int w = 0; w < WIN_; ++w) {
    int pos = t - WCTX_ + w;
    if (pos < 0 || pos >= T_) continue;
    int r = tid + w;  // LDS row: (t - (t0 - WCTX_)) + (w - WCTX_) = tid + w
    float s = 0.f;
    const v8bf* pk = (const v8bf*)(sk + r * DH_);
#pragma unroll
    for (int c = 0; c < 8; ++c) {
      v8bf kk = pk[c];
#pragma unroll
      for (int e = 0; e < 8; ++e) s += qr[c * 8 + e] * (float)kk[e];
    }
    float nm = fmaxf(mx, s);
    float al = __expf(mx - nm);
    float p  = __expf(s - nm);
    lsum     = lsum * al + p;
    const v8bf* pv = (const v8bf*)(sv + r * DH_);
#pragma unroll
    for (int c = 0; c < 8; ++c) {
      v8bf vvv = pv[c];
#pragma unroll
      for (int e = 0; e < 8; ++e)
        o[c * 8 + e] = o[c * 8 + e] * al + p * (float)vvv[e];
    }
    mx = nm;
  }

  float inv = 1.f / lsum;
  bf16* op  = out + ((size_t)(b * T_ + t)) * D_ + h * DH_;
#pragma unroll
  for (int d = 0; d < DH_; ++d) op[d] = f2b(o[d] * inv);
}

// ---------------------------------------------------------------------------
// Host launcher
// ---------------------------------------------------------------------------
extern "C" void kernel_launch(void* const* d_in, const int* in_sizes, int n_in,
                              void* d_out, int out_size, void* d_ws,
                              size_t ws_size, hipStream_t stream) {
  const float* x    = (const float*)d_in[0];
  const float* ln_g = (const float*)d_in[1];
  const float* ln_b = (const float*)d_in[2];
  const float* Wp = (const float*)d_in[3];  const float* bp = (const float*)d_in[4];
  const float* Wq = (const float*)d_in[5];  const float* bq = (const float*)d_in[6];
  const float* Wk = (const float*)d_in[7];  const float* bk = (const float*)d_in[8];
  const float* Wv = (const float*)d_in[9];  const float* bv = (const float*)d_in[10];
  const float* Wo = (const float*)d_in[11]; const float* bo = (const float*)d_in[12];
  const float* Wg = (const float*)d_in[13]; const float* bg = (const float*)d_in[14];

  const int M  = MTOT_;        // 4096
  const int DD = D_ * D_;      // 262144
  const int DG = D_ * 2 * D_;  // 524288

  // Workspace carve-out
  size_t off = 0;
  auto carve = [&](size_t bytes) {
    void* p = (char*)d_ws + off;
    off += (bytes + 255) & ~(size_t)255;
    return p;
  };
  bf16* xnb   = (bf16*)carve((size_t)M * D_ * 2);   // layernorm(x), bf16
  bf16* xb    = (bf16*)carve((size_t)M * D_ * 2);   // x, bf16 (gate concat)
  bf16* qinb  = (bf16*)carve((size_t)M * D_ * 2);   // context_proj(xn), bf16
  float* qf   = (float*)carve((size_t)M * D_ * 4);  // scaled q
  float* kf   = (float*)carve((size_t)M * D_ * 4);
  float* vf   = (float*)carve((size_t)M * D_ * 4);
  bf16* attb  = (bf16*)carve((size_t)M * D_ * 2);   // attention out, bf16
  float* ctxf = (float*)carve((size_t)M * D_ * 4);  // ctx (Wo out), f32
  bf16* ctxb  = (bf16*)carve((size_t)M * D_ * 2);   // ctx, bf16 (gate concat)
  bf16* wpb   = (bf16*)carve((size_t)DD * 2);
  bf16* wqb   = (bf16*)carve((size_t)DD * 2);
  bf16* wkb   = (bf16*)carve((size_t)DD * 2);
  bf16* wvb   = (bf16*)carve((size_t)DD * 2);
  bf16* wob   = (bf16*)carve((size_t)DD * 2);
  bf16* wgb   = (bf16*)carve((size_t)DG * 2);
  (void)ws_size; (void)n_in; (void)in_sizes; (void)out_size;

  // 1) Weights -> bf16
  cvt_kernel<<<(DD + 255) / 256, 256, 0, stream>>>(Wp, wpb, DD);
  cvt_kernel<<<(DD + 255) / 256, 256, 0, stream>>>(Wq, wqb, DD);
  cvt_kernel<<<(DD + 255) / 256, 256, 0, stream>>>(Wk, wkb, DD);
  cvt_kernel<<<(DD + 255) / 256, 256, 0, stream>>>(Wv, wvb, DD);
  cvt_kernel<<<(DD + 255) / 256, 256, 0, stream>>>(Wo, wob, DD);
  cvt_kernel<<<(DG + 255) / 256, 256, 0, stream>>>(Wg, wgb, DG);

  // 2) LayerNorm (+ bf16 copy of x)
  ln_kernel<<<M / 8, 256, 0, stream>>>(x, ln_g, ln_b, xnb, xb);

  const int NOKSPLIT  = 1 << 30;
  const int gemm_grid = (M / 64) * (D_ / 128);  // 64 * 4 = 256 blocks
  const float qscale  = 0.125f;                 // 1/sqrt(DH)

  // 3) q_in = xn @ Wp^T + bp           -> bf16
  gemm_bf16_kernel<EPI_BF16><<<gemm_grid, 256, 0, stream>>>(
      xnb, xnb, NOKSPLIT, D_, wpb, bp, 1.f, M, D_, D_, nullptr, qinb, nullptr, nullptr);
  // 4) q = (q_in @ Wq^T + bq) * scale  -> f32
  gemm_bf16_kernel<EPI_SCALE_F32><<<gemm_grid, 256, 0, stream>>>(
      qinb, qinb, NOKSPLIT, D_, wqb, bq, qscale, M, D_, D_, qf, nullptr, nullptr, nullptr);
  // 5) k = xn @ Wk^T + bk              -> f32
  gemm_bf16_kernel<EPI_F32><<<gemm_grid, 256, 0, stream>>>(
      xnb, xnb, NOKSPLIT, D_, wkb, bk, 1.f, M, D_, D_, kf, nullptr, nullptr, nullptr);
  // 6) v = xn @ Wv^T + bv              -> f32
  gemm_bf16_kernel<EPI_F32><<<gemm_grid, 256, 0, stream>>>(
      xnb, xnb, NOKSPLIT, D_, wvb, bv, 1.f, M, D_, D_, vf, nullptr, nullptr, nullptr);

  // 7) banded attention -> bf16
  dim3 agrid(T_ / ATT_TB, H_, B_);
  attn_kernel<<<agrid, ATT_TB, 0, stream>>>(qf, kf, vf, attb);

  // 8) ctx = att @ Wo^T + bo           -> f32 + bf16
  gemm_bf16_kernel<EPI_BOTH><<<gemm_grid, 256, 0, stream>>>(
      attb, attb, NOKSPLIT, D_, wob, bo, 1.f, M, D_, D_, ctxf, ctxb, nullptr, nullptr);

  // 9) gate = sigmoid([x, ctx] @ Wg^T + bg); out = x*(1-g) + ctx*g
  gemm_bf16_kernel<EPI_GATE><<<gemm_grid, 256, 0, stream>>>(
      xb, ctxb, D_, D_, wgb, bg, 1.f, M, D_, 2 * D_, (float*)d_out, nullptr, x, ctxf);
}
